// Attention_Decoder_78400333021563
// MI455X (gfx1250) — matrix operand order repaired and checked by reading, not compiled, not run on hardware
//
#include <hip/hip_runtime.h>
#include <hip/hip_bf16.h>
#include <math.h>

typedef __attribute__((ext_vector_type(16))) _Float16 v16h;
typedef __attribute__((ext_vector_type(8)))  float    v8f;

#define BS 512
#define TT 1024
#define HH 256

// ---------------------------------------------------------------------------
// Generic f16-WMMA GEMM:  C[M,N] = A[M,K] @ W[N,K]^T + bias[N]
// Block: 256 threads = 8 waves arranged 2(M) x 4(N) -> 32x64 block tile.
// K stepped by 32; A and W staged in LDS as f16 (W transposed to [k][n]).
// ---------------------------------------------------------------------------
#define TM 32
#define TN 64
#define TK 32

__global__ __launch_bounds__(256) void gemm_wmma_kernel(
    const float* __restrict__ A, const float* __restrict__ W,
    const float* __restrict__ bias, float* __restrict__ C,
    int M, int N, int K)
{
  __shared__ _Float16 Al[TM][TK + 8];   // [m][k]
  __shared__ _Float16 Wl[TK][TN + 8];   // [k][n]  (transposed W tile)

  const int tid  = threadIdx.x;
  const int lane = tid & 31;
  const int wave = tid >> 5;
  const int waveM = wave & 1;           // 0..1
  const int waveN = wave >> 1;          // 0..3
  const int mBase  = blockIdx.x * TM;
  const int nBase0 = blockIdx.y * TN;
  const int nBase  = nBase0 + waveN * 16;

  v8f acc = {};

  const int mRow  = waveM * 16 + (lane & 15);   // A row inside block tile
  const int kHalf = (lane >> 4) * 8;            // A K-offset for upper half-wave

  for (int k0 = 0; k0 < K; k0 += TK) {
    // ---- stage A tile (32x32 f32 -> f16), 4 floats/thread, float4 loads ----
    {
      int m  = tid >> 3;
      int kk = (tid & 7) * 4;
      const float4 v = *(const float4*)(A + (size_t)(mBase + m) * K + k0 + kk);
      Al[m][kk + 0] = (_Float16)v.x;
      Al[m][kk + 1] = (_Float16)v.y;
      Al[m][kk + 2] = (_Float16)v.z;
      Al[m][kk + 3] = (_Float16)v.w;
    }
    // ---- stage W tile (64n x 32k), transposed into Wl[k][n] ----
    {
      int n  = tid >> 2;                 // 0..63
      int kk = (tid & 3) * 8;
      int gn = nBase0 + n;
      if (gn < N) {
        const float* src = W + (size_t)gn * K + k0 + kk;
#pragma unroll
        for (int i = 0; i < 8; ++i) Wl[kk + i][n] = (_Float16)src[i];
      } else {
#pragma unroll
        for (int i = 0; i < 8; ++i) Wl[kk + i][n] = (_Float16)0.f;
      }
    }
    __syncthreads();

    // ---- build fragments per ISA lane layout ----
    // A 16x32: lane<16 holds K {0..7,16..23}, lane>=16 holds K {8..15,24..31}
    v16h af = {}, bf = {};
#pragma unroll
    for (int e = 0; e < 16; ++e) {
      int kk = (e & 7) + ((e >> 3) << 4) + kHalf;
      af[e] = Al[mRow][kk];
    }
    // B 32x16: lane L holds column n=L&15, same K set as A
#pragma unroll
    for (int e = 0; e < 16; ++e) bf[e] = Wl[(e & 7) + ((e >> 3) << 4) + kHalf][waveN * 16 + (lane & 15)];

    acc = __builtin_amdgcn_wmma_f32_16x16x32_f16(
        false, af, false, bf, (short)0, acc, false, false);
    __syncthreads();
  }

  // ---- epilogue: D layout: lane l, vgpr r -> M = r + 8*(l>>4), N = l&15 ----
  if (nBase < N) {
    const int n  = nBase + (lane & 15);
    const float bv = bias ? bias[n] : 0.f;
    const int mOff = (lane >> 4) * 8;
#pragma unroll
    for (int r = 0; r < 8; ++r) {
      int m = mBase + waveM * 16 + mOff + r;
      C[(size_t)m * N + n] = acc[r] + bv;
    }
  }
}

// ---------------------------------------------------------------------------
// Fully fused Bahdanau attention, one block per batch row b:
//   pass 1 (WMMA): logit[t] = V_b + sum_h V_w[h]*tanh((enc[b,t,:]@W1^T)[h]
//                                       + W1_b[h] + w2d[b,h])   -> LDS
//   softmax over T in LDS
//   pass 2: cat[b,256+h] = sum_t u[t]*enc[b,t,h]   (enc[b] is L2-hot from
//           pass 1: 1MB/block slice, 192MB device L2)
// A fragments (the wave's 16x256 enc tile) are loaded & converted ONCE per
// segment into 8 register v16h fragments; the nT x k hot loop is then just
// 16 strided f16 LDS reads (B) + 1 v_wmma per iteration.
// ---------------------------------------------------------------------------
#define W1PITCH 264

__global__ __launch_bounds__(256) void attn_fused_kernel(
    const float* __restrict__ enc,   // [bs*T, H]
    const float* __restrict__ W1,    // [H, H]
    const float* __restrict__ W1b,   // [H]
    const float* __restrict__ w2d,   // [bs, H]  (already includes W2_b)
    const float* __restrict__ Vw,    // [H]
    const float* __restrict__ Vb,    // [1]
    float* __restrict__ cat)         // [bs, 512]
{
  extern __shared__ _Float16 W1l[];  // [k][n], pitch W1PITCH : 132 KB
  __shared__ float uLds[TT];         // logits -> softmax weights (4 KB)
  __shared__ float red[8];

  const int tid  = threadIdx.x;
  const int lane = tid & 31;
  const int wave = tid >> 5;
  const int b    = blockIdx.x;

  // stage W1 transposed: global coalesced reads W1[n*256 + k] with k = tid
  for (int n = 0; n < HH; ++n)
    W1l[(size_t)tid * W1PITCH + n] = (_Float16)W1[(size_t)n * HH + tid];
  __syncthreads();

  const float* encB  = enc + (size_t)b * TT * HH;
  const int    kHalf = (lane >> 4) * 8;
  const float  vb    = Vb[0];

  // ---------------- pass 1: logits for all T rows (WMMA) ----------------
  for (int seg = 0; seg < 8; ++seg) {
    const int t0 = seg * 128 + wave * 16;          // this wave's 16-row tile
    const float* encRow = encB + (size_t)(t0 + (lane & 15)) * HH;

    // Hoisted A fragments: whole 16x256 tile in registers (8 x v16h).
    v16h afr[8];
#pragma unroll
    for (int kb = 0; kb < 8; ++kb) {
      const float* ar = encRow + kb * 32 + kHalf;
      float4 c0 = *(const float4*)(ar + 0);
      float4 c1 = *(const float4*)(ar + 4);
      float4 c2 = *(const float4*)(ar + 16);
      float4 c3 = *(const float4*)(ar + 20);
      afr[kb][0]  = (_Float16)c0.x; afr[kb][1]  = (_Float16)c0.y;
      afr[kb][2]  = (_Float16)c0.z; afr[kb][3]  = (_Float16)c0.w;
      afr[kb][4]  = (_Float16)c1.x; afr[kb][5]  = (_Float16)c1.y;
      afr[kb][6]  = (_Float16)c1.z; afr[kb][7]  = (_Float16)c1.w;
      afr[kb][8]  = (_Float16)c2.x; afr[kb][9]  = (_Float16)c2.y;
      afr[kb][10] = (_Float16)c2.z; afr[kb][11] = (_Float16)c2.w;
      afr[kb][12] = (_Float16)c3.x; afr[kb][13] = (_Float16)c3.y;
      afr[kb][14] = (_Float16)c3.z; afr[kb][15] = (_Float16)c3.w;
    }

    float logit[8] = {0.f, 0.f, 0.f, 0.f, 0.f, 0.f, 0.f, 0.f};

    for (int nT = 0; nT < 16; ++nT) {
      v8f acc = {};
#pragma unroll
      for (int kb = 0; kb < 8; ++kb) {
        v16h bf;
        const _Float16* wr = &W1l[(size_t)(kb * 32 + kHalf) * W1PITCH + nT * 16 + (lane & 15)];
#pragma unroll
        for (int e = 0; e < 16; ++e) bf[e] = wr[(size_t)((e & 7) + ((e >> 3) << 4)) * W1PITCH];
        acc = __builtin_amdgcn_wmma_f32_16x16x32_f16(
            false, afr[kb], false, bf, (short)0, acc, false, false);
      }
      // fused epilogue for this 16-col slab: n = nT*16 + (lane&15)
      const int n = nT * 16 + (lane & 15);
      const float add = W1b[n] + w2d[(size_t)b * HH + n];
      const float vw  = Vw[n];
#pragma unroll
      for (int r = 0; r < 8; ++r)
        logit[r] += vw * tanhf(acc[r] + add);
    }

    // reduce over the 16 lanes sharing each row (lane&15)
#pragma unroll
    for (int off = 1; off < 16; off <<= 1)
#pragma unroll
      for (int r = 0; r < 8; ++r)
        logit[r] += __shfl_xor(logit[r], off, 32);

    if ((lane & 15) == 0) {
      const int mOff = (lane >> 4) * 8;
#pragma unroll
      for (int r = 0; r < 8; ++r)
        uLds[t0 + mOff + r] = logit[r] + vb;
    }
  }
  __syncthreads();

  // ---------------- softmax over T, entirely in LDS ----------------
  float l[4];
  float m = -1e30f;
#pragma unroll
  for (int i = 0; i < 4; ++i) { l[i] = uLds[tid + i * 256]; m = fmaxf(m, l[i]); }
#pragma unroll
  for (int off = 16; off >= 1; off >>= 1) m = fmaxf(m, __shfl_xor(m, off, 32));
  if (lane == 0) red[wave] = m;
  __syncthreads();
  float mAll = red[0];
#pragma unroll
  for (int w = 1; w < 8; ++w) mAll = fmaxf(mAll, red[w]);
  __syncthreads();

  float s = 0.f;
#pragma unroll
  for (int i = 0; i < 4; ++i) { l[i] = __expf(l[i] - mAll); s += l[i]; }
#pragma unroll
  for (int off = 16; off >= 1; off >>= 1) s += __shfl_xor(s, off, 32);
  if (lane == 0) red[wave] = s;
  __syncthreads();
  float sAll = 0.f;
#pragma unroll
  for (int w = 0; w < 8; ++w) sAll += red[w];
  const float inv = 1.0f / sAll;
#pragma unroll
  for (int i = 0; i < 4; ++i) uLds[tid + i * 256] = l[i] * inv;
  __syncthreads();

  // ---------------- pass 2: weighted sum (enc[b] L2-resident) ----------------
  const float* e = encB + tid;   // h = tid, coalesced across threads
  float acc = 0.f;
  for (int t = 0; t < TT; ++t) {
    __builtin_prefetch(e + (size_t)(t + 16) * HH, 0, 1);
    acc = fmaf(uLds[t], e[(size_t)t * HH], acc);
  }
  cat[(size_t)b * 512 + HH + tid] = acc;
}

// ---------------------------------------------------------------------------
// GRU gate math: hnew from gi/gh/h.  Optionally aux = hnew + extra.
// ---------------------------------------------------------------------------
__global__ __launch_bounds__(256) void gru_elem_kernel(
    const float* __restrict__ gi, const float* __restrict__ gh,
    const float* __restrict__ h, float* __restrict__ hout,
    const float* __restrict__ extra, float* __restrict__ aux,
    int auxStride, int auxOff)
{
  const int idx = blockIdx.x * 256 + threadIdx.x;   // b*256 + j
  const int b = idx >> 8, j = idx & 255;
  const float* gib = gi + (size_t)b * 768;
  const float* ghb = gh + (size_t)b * 768;
  const float ir = gib[j], iz = gib[256 + j], in_ = gib[512 + j];
  const float hr = ghb[j], hz = ghb[256 + j], hn = ghb[512 + j];
  const float r = 1.f / (1.f + __expf(-(ir + hr)));
  const float z = 1.f / (1.f + __expf(-(iz + hz)));
  const float n = tanhf(in_ + r * hn);
  const float hnew = (1.f - z) * n + z * h[idx];
  hout[idx] = hnew;
  if (aux) {
    const float ev = extra ? extra[idx] : 0.f;
    aux[(size_t)b * auxStride + auxOff + j] = hnew + ev;
  }
}

// ---------------------------------------------------------------------------
extern "C" void kernel_launch(void* const* d_in, const int* in_sizes, int n_in,
                              void* d_out, int out_size, void* d_ws, size_t ws_size,
                              hipStream_t stream) {
  const float* enc    = (const float*)d_in[0];
  const float* dec    = (const float*)d_in[1];   // [512,128]
  const float* attn_h = (const float*)d_in[2];   // [512,256]
  const float* g1h    = (const float*)d_in[3];
  const float* g2h    = (const float*)d_in[4];
  const float* W1w = (const float*)d_in[5];
  const float* W1b = (const float*)d_in[6];
  const float* W2w = (const float*)d_in[7];
  const float* W2b = (const float*)d_in[8];
  const float* Vw  = (const float*)d_in[9];
  const float* Vb  = (const float*)d_in[10];
  const float* projw = (const float*)d_in[11];
  const float* projb = (const float*)d_in[12];
  const float* outw  = (const float*)d_in[13];   // [400,256]
  const float* outb  = (const float*)d_in[14];
  const float* a_wih = (const float*)d_in[15];
  const float* a_whh = (const float*)d_in[16];
  const float* a_bih = (const float*)d_in[17];
  const float* a_bhh = (const float*)d_in[18];
  const float* g1_wih = (const float*)d_in[19];
  const float* g1_whh = (const float*)d_in[20];
  const float* g1_bih = (const float*)d_in[21];
  const float* g1_bhh = (const float*)d_in[22];
  const float* g2_wih = (const float*)d_in[23];
  const float* g2_whh = (const float*)d_in[24];
  const float* g2_bih = (const float*)d_in[25];
  const float* g2_bhh = (const float*)d_in[26];

  float* out_mel = (float*)d_out;              // [512, 400]
  float* out_ah  = out_mel + 512 * 400;        // next_attention_GRU_h
  float* out_h1  = out_ah + 512 * 256;         // next_gru1_h
  float* out_h2  = out_h1 + 512 * 256;         // next_gru2_h

  float* ws     = (float*)d_ws;
  float* gi     = ws;  ws += 512 * 768;
  float* gh     = ws;  ws += 512 * 768;
  float* w2d    = ws;  ws += 512 * 256;
  float* cat    = ws;  ws += 512 * 512;
  float* proj   = ws;  ws += 512 * 256;
  float* g2in   = ws;  ws += 512 * 256;
  float* fin    = ws;  ws += 512 * 256;

  dim3 blk(256);
  auto gemm = [&](const float* A, const float* W, const float* bias, float* C,
                  int M, int N, int K) {
    dim3 grid(M / 32, (N + 63) / 64);
    gemm_wmma_kernel<<<grid, blk, 0, stream>>>(A, W, bias, C, M, N, K);
  };

  // ---- attention GRU (one step) ----
  gemm(dec,    a_wih, a_bih, gi, 512, 768, 128);
  gemm(attn_h, a_whh, a_bhh, gh, 512, 768, 256);
  gru_elem_kernel<<<512, blk, 0, stream>>>(gi, gh, attn_h, out_ah,
                                           nullptr, cat, 512, 0); // cat[:, :256] = d_t

  // ---- w2d = d_t @ W2^T + W2_b ----
  gemm(out_ah, W2w, W2b, w2d, 512, 256, 256);

  // ---- fused attention: WMMA scores + tanh + V-dot + softmax + u.enc ----
  attn_fused_kernel<<<512, blk, HH * W1PITCH * sizeof(_Float16), stream>>>(
      enc, W1w, W1b, w2d, Vw, Vb, cat);                           // cat[:, 256:]

  // ---- proj ----
  gemm(cat, projw, projb, proj, 512, 256, 512);

  // ---- GRU1 ----
  gemm(proj, g1_wih, g1_bih, gi, 512, 768, 256);
  gemm(g1h,  g1_whh, g1_bhh, gh, 512, 768, 256);
  gru_elem_kernel<<<512, blk, 0, stream>>>(gi, gh, g1h, out_h1,
                                           proj, g2in, 256, 0);   // in_gru2

  // ---- GRU2 ----
  gemm(g2in, g2_wih, g2_bih, gi, 512, 768, 256);
  gemm(g2h,  g2_whh, g2_bhh, gh, 512, 768, 256);
  gru_elem_kernel<<<512, blk, 0, stream>>>(gi, gh, g2h, out_h2,
                                           g2in, fin, 256, 0);    // in_gru2+out_gru2

  // ---- output projection ----
  gemm(fin, outw, outb, out_mel, 512, 400, 256);
}
